// GDANet_seg_5119601017105
// MI455X (gfx1250) — matrix operand order repaired and checked by reading, not compiled, not run on hardware
//
#include <hip/hip_runtime.h>

typedef _Float16 h16;
typedef __attribute__((ext_vector_type(16))) _Float16 v16h;
typedef __attribute__((ext_vector_type(8)))  _Float16 v8h;
typedef __attribute__((ext_vector_type(8)))  float    v8f;

#define BNSCL 0.9999950000374997f  /* 1/sqrt(1+1e-5) */

// ------------------------------------------------------------------
// elementwise helpers
// ------------------------------------------------------------------
__global__ void cast_pad(const float* __restrict__ s, h16* __restrict__ d,
                         int M, int K, int Mp, int Kp) {
  int i = blockIdx.x * 256 + threadIdx.x;
  if (i >= Mp * Kp) return;
  int r = i / Kp, c = i % Kp;
  d[i] = (r < M && c < K) ? (h16)s[r * K + c] : (h16)0.f;
}

__global__ void zero_h16(h16* __restrict__ d, int n) {
  int i = blockIdx.x * 256 + threadIdx.x;
  if (i < n) d[i] = (h16)0.f;
}

__global__ void copy_h16(const h16* __restrict__ s, h16* __restrict__ d, int n) {
  int i = blockIdx.x * 256 + threadIdx.x;
  if (i < n) d[i] = s[i];
}

// s:(8,Cs,2048) -> d:(8,Ct,2048) at channel offset `off`
__global__ void copy_bcn(const float* __restrict__ s, float* __restrict__ d,
                         int Cs, int Ct, int off) {
  int i = blockIdx.x * 256 + threadIdx.x;
  int tot = 8 * Cs * 2048;
  if (i >= tot) return;
  int n = i & 2047;
  int c = (i >> 11) % Cs;
  int b = i / (Cs << 11);
  d[((size_t)b * Ct + off + c) * 2048 + n] = s[i];
}

// ------------------------------------------------------------------
// kNN: one block per (query n, batch b); x:(8,C,2048) f32; idx:(8,2048,k)
// ------------------------------------------------------------------
__global__ __launch_bounds__(256)
void knn_kernel(const float* __restrict__ x, int C, int k, int* __restrict__ idx) {
  __shared__ float d2[2048];
  __shared__ float ctr[132];
  __shared__ float rv[256];
  __shared__ int   ri[256];
  const int n = blockIdx.x, b = blockIdx.y, tid = threadIdx.x;
  const float* xb = x + (size_t)b * C * 2048;
  for (int c = tid; c < C; c += 256) ctr[c] = xb[(size_t)c * 2048 + n];
  __syncthreads();
  for (int m = tid; m < 2048; m += 256) {
    float d = 0.f;
    for (int c = 0; c < C; c++) { float t = ctr[c] - xb[(size_t)c * 2048 + m]; d += t * t; }
    d2[m] = d;
  }
  __syncthreads();
  for (int j = 0; j < k; j++) {
    float bv = 3.4e38f; int bi = 0;
    for (int m = tid; m < 2048; m += 256) { float v = d2[m]; if (v < bv) { bv = v; bi = m; } }
    rv[tid] = bv; ri[tid] = bi;
    __syncthreads();
    for (int s = 128; s > 0; s >>= 1) {
      if (tid < s && rv[tid + s] < rv[tid]) { rv[tid] = rv[tid + s]; ri[tid] = ri[tid + s]; }
      __syncthreads();
    }
    if (tid == 0) { idx[((size_t)b * 2048 + n) * k + j] = ri[0]; d2[ri[0]] = 3.4e38f; }
    __syncthreads();
  }
}

// ------------------------------------------------------------------
// edge features: E[Kp][P] f16, P = 8*2048*k, col p = (b*N+n)*k+kk
// rows 0..C-1 = nbr - ctr ; rows C..2C-1 = nbr ; rows >=2C pre-zeroed
// ------------------------------------------------------------------
__global__ void edge_feat(const float* __restrict__ x, const int* __restrict__ idx,
                          int C, int k, h16* __restrict__ E) {
  int p = blockIdx.x * 256 + threadIdx.x;
  int P = 8 * 2048 * k;
  if (p >= P) return;
  int bn = p / k;
  int n = bn & 2047, b = bn >> 11;
  int j = idx[p];
  const float* xb = x + (size_t)b * C * 2048;
  for (int c = 0; c < C; c++) {
    float ctr = xb[(size_t)c * 2048 + n];
    float nb  = xb[(size_t)c * 2048 + j];
    E[(size_t)c * P + p]       = (h16)(nb - ctr);
    E[(size_t)(C + c) * P + p] = (h16)nb;
  }
}

// ------------------------------------------------------------------
// WMMA GEMM (weight stationary conv): Out[M x P] = W[Mp x Kp] * X[Kp x P]
// P is a compile-time constant so B-operand strides fold into immediate
// offsets (no per-iteration 64-bit address chains).
// Kp % 32 == 0, Mp % 16 == 0, P % 16 == 0 : no bounds checks in the loop.
// ------------------------------------------------------------------
template <int P>
__global__ __launch_bounds__(32)
void gemm_conv(const h16* __restrict__ W, const h16* __restrict__ X,
               const float* __restrict__ bias, const float* __restrict__ gam,
               const float* __restrict__ bet, const h16* __restrict__ resid,
               h16* __restrict__ outH, float* __restrict__ outBCN,
               int Mdim, int Kp, int flags, int Ctot, int chOff) {
  const int lane = threadIdx.x & 31;
  const int hf   = lane >> 4;
  const int l16  = lane & 15;
  const int tm = blockIdx.y, tn = blockIdx.x;
  const int rowA = tm * 16 + l16;
  const int col  = tn * 16 + l16;
  const h16* wrow = W + (size_t)rowA * Kp + hf * 16;     // 16B aligned
  const h16* xcol = X + (size_t)(hf * 16) * P + col;
  v8f acc = {0.f, 0.f, 0.f, 0.f, 0.f, 0.f, 0.f, 0.f};
  for (int k0 = 0; k0 < Kp; k0 += 32) {
    v8h a0 = *(const v8h*)(wrow + k0);
    v8h a1 = *(const v8h*)(wrow + k0 + 8);
    v16h a = __builtin_shufflevector(a0, a1, 0, 1, 2, 3, 4, 5, 6, 7,
                                     8, 9, 10, 11, 12, 13, 14, 15);
    const h16* xk = xcol + (size_t)k0 * P;
    v16h bm;
#pragma unroll
    for (int i = 0; i < 16; i++) bm[i] = xk[(size_t)i * P];
    if (k0 + 32 < Kp) __builtin_prefetch(xk + (size_t)32 * P, 0, 1);
    acc = __builtin_amdgcn_wmma_f32_16x16x32_f16(false, a, false, bm, (short)0, acc,
                                                 false, false);
  }
#pragma unroll
  for (int r = 0; r < 8; r++) {
    int m = tm * 16 + hf * 8 + r;
    if (m < Mdim) {
      float v = acc[r];
      if (bias) v += bias[m];
      if (flags & 1) v = v * (BNSCL * gam[m]) + bet[m];
      if (resid) v += (float)resid[(size_t)m * P + col];
      if (flags & 2) v = v > 0.f ? v : 0.f;
      if (outH) outH[(size_t)m * P + col] = (h16)v;
      if (outBCN) {
        int bb = col / 2048, nn = col & 2047;
        outBCN[((size_t)bb * Ctot + chOff + m) * 2048 + nn] = v;
      }
    }
  }
}

// ------------------------------------------------------------------
// attention GEMM (fixed config): att[b](2048 x 512) = TH[b]^T * PH[b] / 512
// TH:[32][8*2048] f16, PH:[32][8*512] f16, ATT:[8][2048][512] f16
// ------------------------------------------------------------------
__global__ __launch_bounds__(32)
void gemm_attn(const h16* __restrict__ TH, const h16* __restrict__ PH,
               h16* __restrict__ ATT) {
  constexpr int Nq = 2048, Mc = 512;
  constexpr size_t BNt = 8 * 2048, BMt = 8 * 512;
  const int lane = threadIdx.x & 31;
  const int hf = lane >> 4, l16 = lane & 15;
  const int tm = blockIdx.y, tn = blockIdx.x, bz = blockIdx.z;
  const h16* tcol = TH + (size_t)bz * Nq + (tm * 16 + l16) + (size_t)(hf * 16) * BNt;
  const h16* pcol = PH + (size_t)bz * Mc + (tn * 16 + l16) + (size_t)(hf * 16) * BMt;
  const int colM = tn * 16 + l16;
  v8f acc = {0.f, 0.f, 0.f, 0.f, 0.f, 0.f, 0.f, 0.f};
  v16h a, bm;
#pragma unroll
  for (int i = 0; i < 8; i++) {
    a[i]     = tcol[(size_t)i * BNt];
    a[i + 8] = tcol[(size_t)(8 + i) * BNt];
  }
#pragma unroll
  for (int i = 0; i < 16; i++) bm[i] = pcol[(size_t)i * BMt];
  acc = __builtin_amdgcn_wmma_f32_16x16x32_f16(false, a, false, bm, (short)0, acc,
                                               false, false);
#pragma unroll
  for (int r = 0; r < 8; r++) {
    int nr = tm * 16 + hf * 8 + r;
    ATT[((size_t)bz * Nq + nr) * Mc + colM] = (h16)(acc[r] * (1.0f / 512.0f));
  }
}

// ------------------------------------------------------------------
// y GEMM (fixed config): Y[b](32 x 2048) = GX[b](32 x 512) * ATT[b]^T
// GX:[32][8*512] f16, ATT:[8][2048][512] f16, Y:[32][8*2048] f16
// ------------------------------------------------------------------
__global__ __launch_bounds__(32)
void gemm_y(const h16* __restrict__ GX, const h16* __restrict__ ATT,
            h16* __restrict__ Y) {
  constexpr int Nq = 2048, Mc = 512;
  constexpr size_t BNt = 8 * 2048, BMt = 8 * 512;
  const int lane = threadIdx.x & 31;
  const int hf = lane >> 4, l16 = lane & 15;
  const int tm = blockIdx.y, tn = blockIdx.x, bz = blockIdx.z;
  const int rowc = tm * 16 + l16;  // channel row < 32
  const int coln = tn * 16 + l16;  // point column
  const h16* grow = GX + (size_t)rowc * BMt + (size_t)bz * Mc + hf * 16;
  const h16* arow = ATT + ((size_t)bz * Nq + coln) * Mc + hf * 16;
  v8f acc = {0.f, 0.f, 0.f, 0.f, 0.f, 0.f, 0.f, 0.f};
#pragma unroll 4
  for (int k0 = 0; k0 < Mc; k0 += 32) {
    v8h a0 = *(const v8h*)(grow + k0);
    v8h a1 = *(const v8h*)(grow + k0 + 8);
    v16h a = __builtin_shufflevector(a0, a1, 0, 1, 2, 3, 4, 5, 6, 7,
                                     8, 9, 10, 11, 12, 13, 14, 15);
    v8h b0 = *(const v8h*)(arow + k0);
    v8h b1 = *(const v8h*)(arow + k0 + 8);
    v16h bm = __builtin_shufflevector(b0, b1, 0, 1, 2, 3, 4, 5, 6, 7,
                                      8, 9, 10, 11, 12, 13, 14, 15);
    acc = __builtin_amdgcn_wmma_f32_16x16x32_f16(false, a, false, bm, (short)0, acc,
                                                 false, false);
  }
#pragma unroll
  for (int r = 0; r < 8; r++) {
    int mr = tm * 16 + hf * 8 + r;
    if (mr < 32) Y[(size_t)mr * BNt + (size_t)bz * Nq + coln] = (h16)acc[r];
  }
}

// ------------------------------------------------------------------
// max over neighbor axis: In [O][B*N*k] -> outH [O][B*N] (+ f32 BCN)
// ------------------------------------------------------------------
__global__ void maxk_kernel(const h16* __restrict__ In, int O, int k,
                            h16* __restrict__ outH, float* __restrict__ outBCN,
                            int Ctot, int chOff) {
  int t = blockIdx.x * 256 + threadIdx.x;
  const int BN = 8 * 2048;
  if (t >= O * BN) return;
  int col = t % BN, o = t / BN;
  size_t P = (size_t)BN * k;
  const h16* src = In + (size_t)o * P + (size_t)col * k;
  float mv = -3.4e38f;
  for (int kk = 0; kk < k; kk++) { float v = (float)src[kk]; mv = v > mv ? v : mv; }
  if (outH) outH[(size_t)o * BN + col] = (h16)mv;
  if (outBCN) {
    int b = col >> 11, n = col & 2047;
    outBCN[((size_t)b * Ctot + chOff + o) * 2048 + n] = mv;
  }
}

// global max over points: xc [512][8*2048] f16 -> xc1 (8,512) f32
__global__ void maxpool_n(const h16* __restrict__ xc, float* __restrict__ xc1) {
  int t = blockIdx.x * 256 + threadIdx.x;
  if (t >= 8 * 512) return;
  int c = t % 512, b = t / 512;
  const h16* src = xc + (size_t)c * (8 * 2048) + b * 2048;
  float mv = -3.4e38f;
  for (int n = 0; n < 2048; n++) { float v = (float)src[n]; mv = v > mv ? v : mv; }
  xc1[t] = mv;
}

// broadcast pooled vector into rows 256..767 of xmid [768][8*2048]
__global__ void broadcast_mid(const float* __restrict__ xc1, h16* __restrict__ xmid) {
  int t = blockIdx.x * 256 + threadIdx.x;
  if (t >= 512 * 8 * 2048) return;
  int col = t % (8 * 2048), c = t / (8 * 2048);
  int b = col >> 11;
  xmid[(size_t)(256 + c) * (8 * 2048) + col] = (h16)xc1[b * 512 + c];
}

// ------------------------------------------------------------------
// GDM: Gaussian-affinity low-pass aggregation + sharp/gentle scores
// ------------------------------------------------------------------
__global__ __launch_bounds__(64)
void gdm_scores(const float* __restrict__ x, const int* __restrict__ idx,
                float* __restrict__ ss, float* __restrict__ sg) {
  int t = blockIdx.x * 64 + threadIdx.x;
  if (t >= 8 * 2048) return;
  int b = t >> 11, n = t & 2047;
  const float* xb = x + (size_t)b * 64 * 2048;
  float ctr[64];
#pragma unroll
  for (int c = 0; c < 64; c++) ctr[c] = xb[c * 2048 + n];
  float w[64]; int jj[64]; float wsum = 0.f;
  for (int kk = 0; kk < 64; kk++) {
    int j = idx[(size_t)t * 64 + kk];
    jj[kk] = j;
    float d = 0.f;
    for (int c = 0; c < 64; c++) { float u = ctr[c] - xb[c * 2048 + j]; d += u * u; }
    float a = __expf(-d * 0.125f);        // sigma = 2 -> 1/(2*sigma^2) = 0.125
    if (a <= 0.2f) a = 0.f;               // tau threshold
    w[kk] = a; wsum += a;
  }
  float inv = 1.f / wsum;
  float s1 = 0.f, s2 = 0.f;
  for (int c = 0; c < 64; c++) {
    float agg = 0.f;
    for (int kk = 0; kk < 64; kk++) agg += w[kk] * xb[c * 2048 + jj[kk]];
    agg *= inv;
    float d1 = ctr[c] - agg; s1 += d1 * d1;
    float d2 = ctr[c] + agg; s2 += d2 * d2;
  }
  ss[t] = s1; sg[t] = s2;
}

// top-Msel (largest) selection per batch: score (8,2048) -> out (8,Msel)
__global__ __launch_bounds__(256)
void top_sel(const float* __restrict__ score, int Msel, int* __restrict__ out) {
  __shared__ float s[2048];
  __shared__ float rv[256];
  __shared__ int   ri[256];
  int b = blockIdx.x, tid = threadIdx.x;
  for (int m = tid; m < 2048; m += 256) s[m] = score[b * 2048 + m];
  __syncthreads();
  for (int j = 0; j < Msel; j++) {
    float bv = -3.4e38f; int bi = 0;
    for (int m = tid; m < 2048; m += 256) { float v = s[m]; if (v > bv) { bv = v; bi = m; } }
    rv[tid] = bv; ri[tid] = bi;
    __syncthreads();
    for (int st = 128; st > 0; st >>= 1) {
      if (tid < st && rv[tid + st] > rv[tid]) { rv[tid] = rv[tid + st]; ri[tid] = ri[tid + st]; }
      __syncthreads();
    }
    if (tid == 0) { out[b * Msel + j] = ri[0]; s[ri[0]] = -3.4e38f; }
    __syncthreads();
  }
}

// gather selected columns: x:(8,C,2048) f32, idxM:(8,Msel) -> out [C][8*Msel] f16
__global__ void gather_cols(const float* __restrict__ x, const int* __restrict__ idxM,
                            h16* __restrict__ out, int C, int Msel) {
  int t = blockIdx.x * 256 + threadIdx.x;
  int tot = 8 * Msel;
  if (t >= tot) return;
  int b = t / Msel;
  int j = idxM[t];
  for (int c = 0; c < C; c++)
    out[(size_t)c * tot + t] = (h16)x[((size_t)b * C + c) * 2048 + j];
}

// ------------------------------------------------------------------
// host orchestration
// ------------------------------------------------------------------
extern "C" void kernel_launch(void* const* d_in, const int* in_sizes, int n_in,
                              void* d_out, int out_size, void* d_ws, size_t ws_size,
                              hipStream_t stream) {
  (void)in_sizes; (void)n_in; (void)out_size; (void)ws_size;
  constexpr int Bb = 8, Np = 2048, Kn = 30, Kg = 64, Ms = 512;
  const int BN = Bb * Np;        // 16384
  const int BM = Bb * Ms;        // 4096
  const int Pk = BN * Kn;        // 491520

  // ---- input pointer mapping (jax pytree leaf order: sorted dict keys) ----
  int t = 0;
  auto F = [&]() { return (const float*)d_in[t++]; };
  struct CBN { const float *b, *be, *g, *w; };
  auto getc = [&]() { CBN c; c.b = F(); c.be = F(); c.g = F(); c.w = F(); return c; };
  CBN c1 = getc(), c11 = getc(), c12 = getc(), c2 = getc(), c21 = getc(), c22 = getc(),
      c3 = getc(), c31 = getc(), c32 = getc(), c4 = getc(), c5 = getc(), c6 = getc(),
      c7 = getc();
  const float* seg_b = F();
  const float* seg_w = F();
  struct SGP { const float *Wb, *Wbe, *Wg, *Ww, *gb, *gw, *phb, *phw, *thb, *thw; };
  auto gets = [&]() {
    SGP s; s.Wb = F(); s.Wbe = F(); s.Wg = F(); s.Ww = F();
    s.gb = F(); s.gw = F(); s.phb = F(); s.phw = F(); s.thb = F(); s.thw = F();
    return s;
  };
  SGP sg1g = gets(), sg1s = gets(), sg2g = gets(), sg2s = gets();
  const float* up_b = F();
  const float* up_w = F();
  const float* points = F();

  // ---- workspace allocator ----
  size_t cur = 0;
  auto alloc = [&](size_t bytes) {
    void* p = (char*)d_ws + cur;
    cur = (cur + bytes + 255) & ~(size_t)255;
    return p;
  };
  // weight cast into zero-padded [Mp][Kp] f16 (Mp%16==0, Kp%32==0)
  auto castw = [&](const float* src, int M, int K) {
    int Mp = (M + 15) & ~15, Kp = (K + 31) & ~31;
    h16* d = (h16*)alloc((size_t)Mp * Kp * 2);
    int n = Mp * Kp;
    cast_pad<<<(n + 255) / 256, 256, 0, stream>>>(src, d, M, K, Mp, Kp);
    return d;
  };

  h16 *c1w = castw(c1.w, 64, 6),     *c11w = castw(c11.w, 64, 64),
      *c12w = castw(c12.w, 64, 128), *c2w = castw(c2.w, 64, 134),
      *c21w = castw(c21.w, 64, 64),  *c22w = castw(c22.w, 64, 128),
      *c3w = castw(c3.w, 128, 262),  *c31w = castw(c31.w, 128, 128),
      *c32w = castw(c32.w, 128, 128), *c4w = castw(c4.w, 512, 256),
      *c5w = castw(c5.w, 256, 768),  *c6w = castw(c6.w, 256, 256),
      *c7w = castw(c7.w, 128, 256),  *segw = castw(seg_w, 50, 300),
      *upw = castw(up_w, 300, 128);
  struct SGH { h16 *Ww, *gw, *phw, *thw; };
  auto casts = [&](const SGP& s) {
    SGH h; h.Ww = castw(s.Ww, 64, 32); h.gw = castw(s.gw, 32, 64);
    h.phw = castw(s.phw, 32, 64); h.thw = castw(s.thw, 32, 64);
    return h;
  };
  SGH h1g = casts(sg1g), h1s = casts(sg1s), h2g = casts(sg2g), h2s = casts(sg2s);

  // intermediates
  int*   idx30 = (int*)alloc((size_t)BN * Kn * 4);
  int*   idx64 = (int*)alloc((size_t)BN * Kg * 4);
  h16*   E  = (h16*)alloc((size_t)288 * Pk * 2);   // max Kp = 288 (block 3)
  h16*   Cb = (h16*)alloc((size_t)128 * Pk * 2);
  h16*   Cc = (h16*)alloc((size_t)128 * Pk * 2);
  float* x1a = (float*)alloc((size_t)64 * BN * 4);
  float* x1n = (float*)alloc((size_t)64 * BN * 4);
  float* x2a = (float*)alloc((size_t)64 * BN * 4);
  float* x2n = (float*)alloc((size_t)64 * BN * 4);
  float* x1t = (float*)alloc((size_t)67 * BN * 4);
  float* x2t = (float*)alloc((size_t)131 * BN * 4);
  h16*   xq1 = (h16*)alloc((size_t)64 * BN * 2);
  h16*   xq2 = (h16*)alloc((size_t)64 * BN * 2);
  float* ssh = (float*)alloc((size_t)BN * 4);
  float* sge = (float*)alloc((size_t)BN * 4);
  int*   idxS = (int*)alloc((size_t)BM * 4);
  int*   idxG = (int*)alloc((size_t)BM * 4);
  h16*   XsS = (h16*)alloc((size_t)64 * BM * 2);
  h16*   XsG = (h16*)alloc((size_t)64 * BM * 2);
  h16*   gx  = (h16*)alloc((size_t)32 * BM * 2);
  h16*   th  = (h16*)alloc((size_t)32 * BN * 2);
  h16*   ph  = (h16*)alloc((size_t)32 * BM * 2);
  h16*   att = (h16*)alloc((size_t)Bb * Np * Ms * 2);
  h16*   yb  = (h16*)alloc((size_t)32 * BN * 2);
  h16*   cat = (h16*)alloc((size_t)128 * BN * 2);
  h16*   xx  = (h16*)alloc((size_t)256 * BN * 2);
  h16*   t3  = (h16*)alloc((size_t)128 * BN * 2);
  h16*   xc  = (h16*)alloc((size_t)512 * BN * 2);
  float* xc1 = (float*)alloc((size_t)Bb * 512 * 4);
  h16*   xmid = (h16*)alloc((size_t)768 * BN * 2);
  h16*   h5 = (h16*)alloc((size_t)256 * BN * 2);
  h16*   h6 = (h16*)alloc((size_t)256 * BN * 2);
  h16*   h7 = (h16*)alloc((size_t)128 * BN * 2);
  h16*   hup = (h16*)alloc((size_t)320 * BN * 2);  // 300 rows + pad to 320

  auto zero = [&](h16* p, int n) {
    zero_h16<<<(n + 255) / 256, 256, 0, stream>>>(p, n);
  };
  auto conv = [&](const h16* W, const h16* X, int Mo, int K, int P,
                  const float* bias, const float* g, const float* be, int flags,
                  const h16* res, h16* oH, float* oB, int Ct, int off) {
    int Kp = (K + 31) & ~31;
    dim3 gr(P / 16, (Mo + 15) / 16, 1);
    if (P == 491520)
      gemm_conv<491520><<<gr, 32, 0, stream>>>(W, X, bias, g, be, res, oH, oB,
                                               Mo, Kp, flags, Ct, off);
    else if (P == 16384)
      gemm_conv<16384><<<gr, 32, 0, stream>>>(W, X, bias, g, be, res, oH, oB,
                                              Mo, Kp, flags, Ct, off);
    else
      gemm_conv<4096><<<gr, 32, 0, stream>>>(W, X, bias, g, be, res, oH, oB,
                                             Mo, Kp, flags, Ct, off);
  };
  auto sgcam = [&](const h16* Xq, const SGP& p, const SGH& w, const h16* Xs,
                   h16* outRows) {
    conv(w.gw, Xs, 32, 64, BM, p.gb, nullptr, nullptr, 0, nullptr, gx, nullptr, 0, 0);
    conv(w.thw, Xq, 32, 64, BN, p.thb, nullptr, nullptr, 0, nullptr, th, nullptr, 0, 0);
    conv(w.phw, Xs, 32, 64, BM, p.phb, nullptr, nullptr, 0, nullptr, ph, nullptr, 0, 0);
    gemm_attn<<<dim3(Ms / 16, Np / 16, Bb), 32, 0, stream>>>(th, ph, att);
    gemm_y<<<dim3(Np / 16, 2, Bb), 32, 0, stream>>>(gx, att, yb);
    conv(w.Ww, yb, 64, 32, BN, p.Wb, p.Wg, p.Wbe, 1 /*bn*/, Xq, outRows, nullptr, 0, 0);
  };
  auto gdm = [&](const float* xa) {
    knn_kernel<<<dim3(Np, Bb), 256, 0, stream>>>(xa, 64, Kg, idx64);
    gdm_scores<<<(BN + 63) / 64, 64, 0, stream>>>(xa, idx64, ssh, sge);
    top_sel<<<Bb, 256, 0, stream>>>(ssh, Ms, idxS);
    top_sel<<<Bb, 256, 0, stream>>>(sge, Ms, idxG);
    gather_cols<<<(BM + 255) / 256, 256, 0, stream>>>(xa, idxS, XsS, 64, Ms);
    gather_cols<<<(BM + 255) / 256, 256, 0, stream>>>(xa, idxG, XsG, 64, Ms);
  };

  // ================= block 1 =================
  knn_kernel<<<dim3(Np, Bb), 256, 0, stream>>>(points, 3, Kn, idx30);
  zero(E + (size_t)6 * Pk, (32 - 6) * Pk);        // pad rows 6..31
  edge_feat<<<(Pk + 255) / 256, 256, 0, stream>>>(points, idx30, 3, Kn, E);
  conv(c1w, E, 64, 6, Pk, c1.b, c1.g, c1.be, 3, nullptr, Cb, nullptr, 0, 0);
  conv(c11w, Cb, 64, 64, Pk, c11.b, c11.g, c11.be, 3, nullptr, Cc, nullptr, 0, 0);
  maxk_kernel<<<(64 * BN + 255) / 256, 256, 0, stream>>>(Cc, 64, Kn, xq1, x1a, 64, 0);
  gdm(x1a);
  sgcam(xq1, sg1s, h1s, XsS, cat);
  sgcam(xq1, sg1g, h1g, XsG, cat + (size_t)64 * BN);
  conv(c12w, cat, 64, 128, BN, c12.b, c12.g, c12.be, 3, nullptr, xx, x1n, 64, 0);
  copy_bcn<<<(3 * BN + 255) / 256, 256, 0, stream>>>(points, x1t, 3, 67, 0);
  copy_bcn<<<(64 * BN + 255) / 256, 256, 0, stream>>>(x1n, x1t, 64, 67, 3);

  // ================= block 2 =================
  knn_kernel<<<dim3(Np, Bb), 256, 0, stream>>>(x1t, 67, Kn, idx30);
  zero(E + (size_t)134 * Pk, (160 - 134) * Pk);   // pad rows 134..159
  edge_feat<<<(Pk + 255) / 256, 256, 0, stream>>>(x1t, idx30, 67, Kn, E);
  conv(c2w, E, 64, 134, Pk, c2.b, c2.g, c2.be, 3, nullptr, Cb, nullptr, 0, 0);
  conv(c21w, Cb, 64, 64, Pk, c21.b, c21.g, c21.be, 3, nullptr, Cc, nullptr, 0, 0);
  maxk_kernel<<<(64 * BN + 255) / 256, 256, 0, stream>>>(Cc, 64, Kn, xq2, x2a, 64, 0);
  gdm(x2a);
  sgcam(xq2, sg2s, h2s, XsS, cat);
  sgcam(xq2, sg2g, h2g, XsG, cat + (size_t)64 * BN);
  conv(c22w, cat, 64, 128, BN, c22.b, c22.g, c22.be, 3, nullptr,
       xx + (size_t)64 * BN, x2n, 64, 0);
  copy_bcn<<<(67 * BN + 255) / 256, 256, 0, stream>>>(x1t, x2t, 67, 131, 0);
  copy_bcn<<<(64 * BN + 255) / 256, 256, 0, stream>>>(x2n, x2t, 64, 131, 67);

  // ================= block 3 =================
  knn_kernel<<<dim3(Np, Bb), 256, 0, stream>>>(x2t, 131, Kn, idx30);
  zero(E + (size_t)262 * Pk, (288 - 262) * Pk);   // pad rows 262..287
  edge_feat<<<(Pk + 255) / 256, 256, 0, stream>>>(x2t, idx30, 131, Kn, E);
  conv(c3w, E, 128, 262, Pk, c3.b, c3.g, c3.be, 3, nullptr, Cb, nullptr, 0, 0);
  conv(c31w, Cb, 128, 128, Pk, c31.b, c31.g, c31.be, 3, nullptr, Cc, nullptr, 0, 0);
  maxk_kernel<<<(128 * BN + 255) / 256, 256, 0, stream>>>(Cc, 128, Kn, t3, nullptr, 0, 0);
  conv(c32w, t3, 128, 128, BN, c32.b, c32.g, c32.be, 3, nullptr,
       xx + (size_t)128 * BN, nullptr, 0, 0);

  // ================= head =================
  conv(c4w, xx, 512, 256, BN, c4.b, c4.g, c4.be, 3, nullptr, xc, nullptr, 0, 0);
  maxpool_n<<<(Bb * 512 + 255) / 256, 256, 0, stream>>>(xc, xc1);
  copy_h16<<<(256 * BN + 255) / 256, 256, 0, stream>>>(xx, xmid, 256 * BN);
  broadcast_mid<<<(512 * BN + 255) / 256, 256, 0, stream>>>(xc1, xmid);
  conv(c5w, xmid, 256, 768, BN, c5.b, c5.g, c5.be, 3, nullptr, h5, nullptr, 0, 0);
  conv(c6w, h5, 256, 256, BN, c6.b, c6.g, c6.be, 3, nullptr, h6, nullptr, 0, 0);
  conv(c7w, h6, 128, 256, BN, c7.b, c7.g, c7.be, 3, nullptr, h7, nullptr, 0, 0);
  zero(hup + (size_t)300 * BN, (320 - 300) * BN); // pad rows 300..319
  conv(upw, h7, 300, 128, BN, up_b, nullptr, nullptr, 0, nullptr, hup, nullptr, 0, 0);
  conv(segw, hup, 50, 300, BN, seg_b, nullptr, nullptr, 0, nullptr, nullptr,
       (float*)d_out, 50, 0);
}
